// qS5SSM_42030549958944
// MI455X (gfx1250) — matrix operand order repaired and checked by reading, not compiled, not run on hardware
//
#include <hip/hip_runtime.h>
#include <hip/hip_bf16.h>

#define LSEQ 32768
#define HDIM 512
#define PDIM 256
#define NCH  128
#define CLEN 256   /* LSEQ / NCH */
#define SUBL 64    /* timesteps per TDM sub-chunk */
#define NSUB (CLEN / SUBL)

typedef __bf16 bf16_t;
typedef __attribute__((ext_vector_type(16))) __bf16 v16bf;
typedef __attribute__((ext_vector_type(8)))  __bf16 v8bf;
typedef __attribute__((ext_vector_type(4)))  __bf16 v4bf;
typedef __attribute__((ext_vector_type(8)))  float  v8f;
typedef __attribute__((ext_vector_type(4)))  unsigned int v4u;
typedef __attribute__((ext_vector_type(4)))  int v4i;
typedef __attribute__((ext_vector_type(8)))  int v8i;

struct alignas(8) cplx { float re, im; };

__device__ __forceinline__ v16bf cat8(v8bf lo, v8bf hi) {
  return __builtin_shufflevector(lo, hi, 0,1,2,3,4,5,6,7,8,9,10,11,12,13,14,15);
}

// ---- Tensor Data Mover: 2D tile load Global->LDS (D# per CDNA5 ISA §8.3/8.4)
// tile_d0 contiguous elements per row, tile_d1 rows, element size 8B (float2),
// row pitch d0_stride elements. Groups 2/3 zero (<=2D tensor).
__device__ __forceinline__ void tdm_load_2d(unsigned long long gaddr, unsigned int lds_addr,
                                            unsigned int tile_d0, unsigned int tile_d1,
                                            unsigned int tens_d0, unsigned int tens_d1,
                                            unsigned int d0_stride) {
  v4u g0;
  g0[0] = 1u;                                            // count=1, no gather
  g0[1] = lds_addr;                                      // lds_addr[31:0]
  g0[2] = (unsigned int)(gaddr & 0xffffffffu);           // global_addr[31:0]
  g0[3] = (unsigned int)((gaddr >> 32) & 0x01ffffffu)    // global_addr[56:32]
        | (2u << 30);                                    // type=2 ("image")
  v8i g1;
  g1[0] = (int)(3u << 16);                               // wg_mask=0, data_size=3 (8B)
  g1[1] = (int)((tens_d0 & 0xffffu) << 16);              // tensor_dim0[15:0]
  g1[2] = (int)((tens_d0 >> 16) | ((tens_d1 & 0xffffu) << 16)); // t0 hi | t1 lo
  g1[3] = (int)((tens_d1 >> 16) | ((tile_d0 & 0xffffu) << 16)); // t1 hi | tile_dim0
  g1[4] = (int)(tile_d1 & 0xffffu);                      // tile_dim1, tile_dim2=0
  g1[5] = (int)d0_stride;                                // tensor_dim0_stride[31:0]
  g1[6] = 0;                                             // stride hi, dim1_stride lo
  g1[7] = 0;
  v4i z4 = {0, 0, 0, 0};
#if __clang_major__ >= 23
  v8i z8 = {0, 0, 0, 0, 0, 0, 0, 0};
  __builtin_amdgcn_tensor_load_to_lds(g0, g1, z4, z4, z8, 0);
#else
  __builtin_amdgcn_tensor_load_to_lds(g0, g1, z4, z4, 0);
#endif
}

// ---------------------------------------------------------------- setup (P)
__global__ void k_setup(const float* __restrict__ lre, const float* __restrict__ lim,
                        const float* __restrict__ lstep,
                        cplx* __restrict__ lam, cplx* __restrict__ coef) {
  int p = threadIdx.x;               // blockDim == PDIM
  float step = __expf(lstep[p]);
  float ar = lre[p], ai = lim[p];
  float mag = __expf(ar * step);
  float s, c;
  __sincosf(ai * step, &s, &c);
  float lbre = mag * c, lbim = mag * s;
  lam[p] = {lbre, lbim};
  float den = ar * ar + ai * ai;
  float nre = lbre - 1.0f, nim = lbim;
  coef[p] = {(nre * ar + nim * ai) / den, (nim * ar - nre * ai) / den};
}

// ---------------------------------------------- B_bar = coef[p] * B_tilde[p,h]
__global__ void k_pack_b(const float* __restrict__ B, const cplx* __restrict__ coef,
                         bf16_t* __restrict__ bre, bf16_t* __restrict__ bim) {
  int idx = blockIdx.x * 256 + threadIdx.x;     // idx = p*HDIM + h
  cplx cf = coef[idx >> 9];                     // HDIM == 512
  float br = B[2 * idx], bi = B[2 * idx + 1];
  bre[idx] = (bf16_t)(cf.re * br - cf.im * bi);
  bim[idx] = (bf16_t)(cf.re * bi + cf.im * br);
}

// ------------------------------------------------ C -> bf16 (re, -im) (H,P)
__global__ void k_pack_c(const float* __restrict__ C,
                         bf16_t* __restrict__ cre, bf16_t* __restrict__ cimn) {
  int idx = blockIdx.x * 256 + threadIdx.x;     // idx = h*PDIM + p
  cre[idx]  = (bf16_t)C[2 * idx];
  cimn[idx] = (bf16_t)(-C[2 * idx + 1]);
}

// ------------------------------------------------------------- u -> bf16
__global__ void k_pack_u(const float* __restrict__ u, bf16_t* __restrict__ ub) {
  int i = (blockIdx.x * 256 + threadIdx.x) * 4;
  float4 v = *(const float4*)(u + i);
  v4bf o = {(bf16_t)v.x, (bf16_t)v.y, (bf16_t)v.z, (bf16_t)v.w};
  *(v4bf*)(ub + i) = o;
}

// --------------------------------- GEMM1: Bu[l,p] = sum_h u[l,h] * Bbar[p,h]
__global__ void k_gemm1(const bf16_t* __restrict__ ub,
                        const bf16_t* __restrict__ bre,
                        const bf16_t* __restrict__ bim,
                        cplx* __restrict__ Bu) {
  int lane = threadIdx.x & 31;
  int wave = (blockIdx.x << 3) + (threadIdx.x >> 5);
  int tm = wave >> 4;                 // PDIM/16 = 16 tiles along N
  int tn = wave & 15;
  int l0 = tm * 16, p0 = tn * 16;
  int n  = lane & 15, hi = lane >> 4;

  const bf16_t* arow = ub  + (size_t)(l0 + n) * HDIM;           // A row (M=n)
  const bf16_t* brow = bre + (size_t)(p0 + n) * HDIM + hi * 16; // B col (N=n)
  const bf16_t* irow = bim + (size_t)(p0 + n) * HDIM + hi * 16;

  v8f cr = {}, ci = {};
  for (int k0 = 0; k0 < HDIM; k0 += 32) {
    v16bf a = cat8(*(const v8bf*)(arow + k0 + hi * 16),
                   *(const v8bf*)(arow + k0 + hi * 16 + 8));
    v16bf b0 = *(const v16bf*)(brow + k0);
    v16bf b1 = *(const v16bf*)(irow + k0);
    cr = __builtin_amdgcn_wmma_f32_16x16x32_bf16(false, a, false, b0, (short)0, cr, false, false);
    ci = __builtin_amdgcn_wmma_f32_16x16x32_bf16(false, a, false, b1, (short)0, ci, false, false);
  }
  int col = p0 + n;
  #pragma unroll
  for (int r = 0; r < 8; ++r) {
    int row = l0 + r + (hi << 3);
    Bu[(size_t)row * PDIM + col] = {cr[r], ci[r]};
  }
}

// --------------------------- scan pass 1: per-chunk final state (zero init)
// TDM double-buffers 64x256 float2 sub-chunks Global->LDS (s_wait_tensorcnt),
// all 8 waves consume the staged sub-chunk via LDS.
__global__ void k_scan1(const cplx* __restrict__ Bu, const cplx* __restrict__ lam,
                        cplx* __restrict__ finals) {
  extern __shared__ cplx sbuf[];                 // 2 * SUBL * PDIM
  int p = threadIdx.x;
  int c = blockIdx.x;
  bool issuer = (threadIdx.x >> 5) == 0;
  unsigned long long gbase = (unsigned long long)(const void*)Bu
                           + (unsigned long long)c * CLEN * PDIM * sizeof(cplx);
  unsigned int lds0 = (unsigned int)(unsigned long long)(const void*)sbuf;
  const unsigned int subbytes = SUBL * PDIM * sizeof(cplx);

  if (issuer) tdm_load_2d(gbase, lds0, PDIM, SUBL, PDIM, SUBL, PDIM);

  cplx a = lam[p];
  float xr = 0.f, xi = 0.f;
  #pragma unroll
  for (int s = 0; s < NSUB; ++s) {
    if (issuer) {
      if (s + 1 < NSUB)
        tdm_load_2d(gbase + (unsigned long long)(s + 1) * subbytes,
                    lds0 + ((s + 1) & 1) * subbytes, PDIM, SUBL, PDIM, SUBL, PDIM);
      if (s + 1 < NSUB) __builtin_amdgcn_s_wait_tensorcnt(1);
      else              __builtin_amdgcn_s_wait_tensorcnt(0);
    }
    __syncthreads();
    const cplx* rows = sbuf + (s & 1) * (SUBL * PDIM) + p;
    for (int y = 0; y < SUBL; ++y) {
      cplx b = rows[(size_t)y * PDIM];
      float nr = fmaf(a.re, xr, fmaf(-a.im, xi, b.re));
      float ni = fmaf(a.re, xi, fmaf( a.im, xr, b.im));
      xr = nr; xi = ni;
    }
    __syncthreads();
  }
  finals[c * PDIM + p] = {xr, xi};
}

// ------------- scan pass 2: exclusive prefix of chunk carries, a^CLEN decay
__global__ void k_scan2(const cplx* __restrict__ finals, const cplx* __restrict__ lam,
                        cplx* __restrict__ carry) {
  int p = threadIdx.x;
  cplx a = lam[p];
  float ar = a.re, ai = a.im;
  #pragma unroll
  for (int i = 0; i < 8; ++i) {       // a^(2^8) = a^CLEN
    float nr = ar * ar - ai * ai;
    float ni = 2.f * ar * ai;
    ar = nr; ai = ni;
  }
  float fr = 0.f, fi = 0.f;
  for (int c = 0; c < NCH; ++c) {
    carry[c * PDIM + p] = {fr, fi};
    cplx s = finals[c * PDIM + p];
    float nr = fmaf(ar, fr, fmaf(-ai, fi, s.re));
    float ni = fmaf(ar, fi, fmaf( ai, fr, s.im));
    fr = nr; fi = ni;
  }
}

// ------------------- scan pass 3: rescan with true carry, emit xs as bf16
__global__ void k_scan3(const cplx* __restrict__ Bu, const cplx* __restrict__ lam,
                        const cplx* __restrict__ carry,
                        bf16_t* __restrict__ xro, bf16_t* __restrict__ xio) {
  extern __shared__ cplx sbuf[];                 // 2 * SUBL * PDIM
  int p = threadIdx.x;
  int c = blockIdx.x;
  bool issuer = (threadIdx.x >> 5) == 0;
  unsigned long long gbase = (unsigned long long)(const void*)Bu
                           + (unsigned long long)c * CLEN * PDIM * sizeof(cplx);
  unsigned int lds0 = (unsigned int)(unsigned long long)(const void*)sbuf;
  const unsigned int subbytes = SUBL * PDIM * sizeof(cplx);

  if (issuer) tdm_load_2d(gbase, lds0, PDIM, SUBL, PDIM, SUBL, PDIM);

  cplx a = lam[p];
  cplx f = carry[c * PDIM + p];
  float xr = f.re, xi = f.im;
  size_t ob = (size_t)c * CLEN * PDIM + p;
  #pragma unroll
  for (int s = 0; s < NSUB; ++s) {
    if (issuer) {
      if (s + 1 < NSUB)
        tdm_load_2d(gbase + (unsigned long long)(s + 1) * subbytes,
                    lds0 + ((s + 1) & 1) * subbytes, PDIM, SUBL, PDIM, SUBL, PDIM);
      if (s + 1 < NSUB) __builtin_amdgcn_s_wait_tensorcnt(1);
      else              __builtin_amdgcn_s_wait_tensorcnt(0);
    }
    __syncthreads();
    const cplx* rows = sbuf + (s & 1) * (SUBL * PDIM) + p;
    size_t obs = ob + (size_t)s * SUBL * PDIM;
    for (int y = 0; y < SUBL; ++y) {
      cplx b = rows[(size_t)y * PDIM];
      float nr = fmaf(a.re, xr, fmaf(-a.im, xi, b.re));
      float ni = fmaf(a.re, xi, fmaf( a.im, xr, b.im));
      xr = nr; xi = ni;
      xro[obs + (size_t)y * PDIM] = (bf16_t)xr;
      xio[obs + (size_t)y * PDIM] = (bf16_t)xi;
    }
    __syncthreads();
  }
}

// -- GEMM2: out[l,h] = 2*(xs_re·Cre - xs_im·Cim)[l,h] + D[h]*u[l,h]
__global__ void k_gemm2(const bf16_t* __restrict__ xr, const bf16_t* __restrict__ xi,
                        const bf16_t* __restrict__ cre, const bf16_t* __restrict__ cimn,
                        const float* __restrict__ D, const float* __restrict__ u,
                        float* __restrict__ out) {
  int lane = threadIdx.x & 31;
  int wave = (blockIdx.x << 3) + (threadIdx.x >> 5);
  int tm = wave >> 5;                 // HDIM/16 = 32 tiles along N
  int tn = wave & 31;
  int l0 = tm * 16, h0 = tn * 16;
  int n = lane & 15, hi = lane >> 4;

  const bf16_t* ar = xr + (size_t)(l0 + n) * PDIM;
  const bf16_t* ai = xi + (size_t)(l0 + n) * PDIM;
  const bf16_t* b0 = cre  + (size_t)(h0 + n) * PDIM + hi * 16;
  const bf16_t* b1 = cimn + (size_t)(h0 + n) * PDIM + hi * 16;

  v8f acc = {};
  for (int k0 = 0; k0 < PDIM; k0 += 32) {
    v16bf Ar = cat8(*(const v8bf*)(ar + k0 + hi * 16),
                    *(const v8bf*)(ar + k0 + hi * 16 + 8));
    v16bf Ai = cat8(*(const v8bf*)(ai + k0 + hi * 16),
                    *(const v8bf*)(ai + k0 + hi * 16 + 8));
    v16bf Br = *(const v16bf*)(b0 + k0);
    v16bf Bi = *(const v16bf*)(b1 + k0);
    acc = __builtin_amdgcn_wmma_f32_16x16x32_bf16(false, Ar, false, Br, (short)0, acc, false, false);
    acc = __builtin_amdgcn_wmma_f32_16x16x32_bf16(false, Ai, false, Bi, (short)0, acc, false, false);
  }
  int col = h0 + n;
  float d = D[col];
  #pragma unroll
  for (int r = 0; r < 8; ++r) {
    size_t idx = (size_t)(l0 + r + (hi << 3)) * HDIM + col;
    out[idx] = 2.0f * acc[r] + d * u[idx];
  }
}

extern "C" void kernel_launch(void* const* d_in, const int* in_sizes, int n_in,
                              void* d_out, int out_size, void* d_ws, size_t ws_size,
                              hipStream_t stream) {
  const float* Lre   = (const float*)d_in[0];
  const float* Lim   = (const float*)d_in[1];
  const float* B     = (const float*)d_in[2];
  const float* C     = (const float*)d_in[3];
  const float* D     = (const float*)d_in[4];
  const float* lstep = (const float*)d_in[5];
  const float* u     = (const float*)d_in[6];
  float* out = (float*)d_out;

  char* ws = (char*)d_ws;
  size_t off = 0;
  auto alloc = [&](size_t bytes) { void* p = ws + off; off = (off + bytes + 255) & ~(size_t)255; return p; };

  cplx*   lam    = (cplx*)  alloc(PDIM * sizeof(cplx));
  cplx*   coef   = (cplx*)  alloc(PDIM * sizeof(cplx));
  bf16_t* bre    = (bf16_t*)alloc((size_t)PDIM * HDIM * 2);
  bf16_t* bim    = (bf16_t*)alloc((size_t)PDIM * HDIM * 2);
  bf16_t* cre    = (bf16_t*)alloc((size_t)HDIM * PDIM * 2);
  bf16_t* cimn   = (bf16_t*)alloc((size_t)HDIM * PDIM * 2);
  bf16_t* ub     = (bf16_t*)alloc((size_t)LSEQ * HDIM * 2);
  cplx*   Bu     = (cplx*)  alloc((size_t)LSEQ * PDIM * sizeof(cplx));
  bf16_t* xsre   = (bf16_t*)alloc((size_t)LSEQ * PDIM * 2);
  bf16_t* xsim   = (bf16_t*)alloc((size_t)LSEQ * PDIM * 2);
  cplx*   finals = (cplx*)  alloc((size_t)NCH * PDIM * sizeof(cplx));
  cplx*   carry  = (cplx*)  alloc((size_t)NCH * PDIM * sizeof(cplx));

  const size_t scan_lds = (size_t)2 * SUBL * PDIM * sizeof(cplx);  // 256 KB of 320 KB WGP LDS

  k_setup <<<1, PDIM, 0, stream>>>(Lre, Lim, lstep, lam, coef);
  k_pack_b<<<(PDIM * HDIM) / 256, 256, 0, stream>>>(B, coef, bre, bim);
  k_pack_c<<<(HDIM * PDIM) / 256, 256, 0, stream>>>(C, cre, cimn);
  k_pack_u<<<(LSEQ * HDIM) / 1024, 256, 0, stream>>>(u, ub);
  k_gemm1 <<<(LSEQ / 16) * (PDIM / 16) / 8, 256, 0, stream>>>(ub, bre, bim, Bu);
  k_scan1 <<<NCH, PDIM, scan_lds, stream>>>(Bu, lam, finals);
  k_scan2 <<<1, PDIM, 0, stream>>>(finals, lam, carry);
  k_scan3 <<<NCH, PDIM, scan_lds, stream>>>(Bu, lam, carry, xsre, xsim);
  k_gemm2 <<<(LSEQ / 16) * (HDIM / 16) / 8, 256, 0, stream>>>(xsre, xsim, cre, cimn, D, u, out);
}
